// Attention_50646254355297
// MI455X (gfx1250) — compile-verified, matrix operand order checked by reading, not run on hardware
//
#include <hip/hip_runtime.h>

// ---------------------------------------------------------------------------
// Problem constants:  B=2, T=2048, HID=2048, NH=16, NKV=4, HD=128, GROUPS=4.
// ---------------------------------------------------------------------------
constexpr int CB   = 2;
constexpr int CT   = 2048;
constexpr int CHID = 2048;
constexpr int CNH  = 16;
constexpr int CNKV = 4;
constexpr int CHD  = 128;
constexpr int CM   = CB * CT;        // 4096
constexpr int CKVD = CNKV * CHD;     // 512

typedef __attribute__((ext_vector_type(16))) __bf16    v16bf;
typedef __attribute__((ext_vector_type(8)))  float     v8f;
typedef __attribute__((ext_vector_type(4)))  unsigned  u32x4;
typedef __attribute__((ext_vector_type(8)))  int       i32x8;
typedef __attribute__((ext_vector_type(4)))  int       i32x4;

#if defined(__has_builtin)
#if __has_builtin(__builtin_amdgcn_tensor_load_to_lds)
#define HAVE_TDM 1
#endif
#endif

// ---------------------------------------------------------------------------
// TDM: DMA a 2D tile (rows x tensorW8 8-byte units, row stride stride8 units)
// from global memory into LDS at byte offset ldsOff.  D# bitfields per
// CDNA5 ISA.  Issued by one wave; EXEC ignored.
// ---------------------------------------------------------------------------
#ifdef HAVE_TDM
__device__ __forceinline__ void tdm_load_2d_to_lds(unsigned ldsOff, const void* gptr,
                                                   unsigned tensorW8, unsigned rows,
                                                   unsigned stride8) {
    unsigned long long ga = (unsigned long long)(size_t)gptr;
    u32x4 g0 = {};
    g0[0] = 1u;                                      // count=1, user mode
    g0[1] = ldsOff;                                  // lds_addr (bytes)
    g0[2] = (unsigned)ga;                            // global_addr[31:0]
    g0[3] = (unsigned)((ga >> 32) & 0x01ffffffu) | (2u << 30);  // [56:32] | type=2
    i32x8 g1 = {};
    g1[0] = (int)(3u << 16);                         // data_size = 8 bytes
    g1[1] = (int)((tensorW8 & 0xffffu) << 16);       // tensor_dim0[15:0] @ bit48
    g1[2] = (int)(((tensorW8 >> 16) & 0xffffu) | ((rows & 0xffffu) << 16));
    g1[3] = (int)(((rows >> 16) & 0xffffu) | ((tensorW8 & 0xffffu) << 16)); // tile_dim0
    g1[4] = (int)(rows & 0xffffu);                   // tile_dim1 (tile_dim2 = 0)
    g1[5] = (int)stride8;                            // tensor_dim0_stride[31:0]
    g1[6] = 0;
    g1[7] = 0;
    i32x4 z4 = {};
#if __clang_major__ >= 23
    i32x8 z8 = {};
    __builtin_amdgcn_tensor_load_to_lds(g0, g1, z4, z4, z8, 0);
#else
    __builtin_amdgcn_tensor_load_to_lds(g0, g1, z4, z4, 0);
#endif
}
#endif

// ---------------------------------------------------------------------------
// Fragment loader: ISA 16-bit A-matrix 16x32 layout (wave32).
//   lane l: row = l&15, half = l>>4
//   VGPR v in 0..3 -> K = 8*half + 2v,+1 ; v in 4..7 -> K = 16 + 8*half + ...
// Same loader serves B when B is staged transposed (rows indexed by N).
// ---------------------------------------------------------------------------
__device__ __forceinline__ v16bf load_frag(const __bf16* base, int stride, int lane) {
    const int row = lane & 15;
    const int lh  = lane >> 4;
    const __bf16* r = base + (size_t)row * stride + lh * 8;
    union { v16bf v; unsigned u[8]; } f;
#pragma unroll
    for (int i = 0; i < 4; ++i) {
        f.u[i]     = *(const unsigned*)(r + 2 * i);
        f.u[i + 4] = *(const unsigned*)(r + 16 + 2 * i);
    }
    return f.v;
}

// ---------------------------------------------------------------------------
// Elementwise f32 -> bf16
// ---------------------------------------------------------------------------
__global__ void f32_to_bf16_kernel(const float* __restrict__ in,
                                   __bf16* __restrict__ out, int n) {
    int i = blockIdx.x * blockDim.x + threadIdx.x;
    if (i < n) out[i] = (__bf16)in[i];
}

// ---------------------------------------------------------------------------
// Tiled bf16 GEMM, f32 out.  C[M,N] = A[M,K] @ B[K,N].  N,K compile-time.
// Block 256 thr = 8 waves; tile 128x128, K-step 32; wave grid 2x4 (64x32 each).
// DOUBLE BUFFERED: tile i+1's A-tile DMA (tensor_load_to_lds) and B-tile
// global loads are issued before computing tile i, so TDM + memory latency
// hide under the 8 WMMAs/wave of the current step.  B tile is stored
// transposed ([n][k]) with packed k-pair b32 stores after the compute phase.
// ---------------------------------------------------------------------------
#define GTM 128
#define GTN 128
#define GTK 32

template<int N, int K>
__global__ __launch_bounds__(256)
void gemm_bf16_kernel(const __bf16* __restrict__ A, const __bf16* __restrict__ B,
                      float* __restrict__ C) {
    __shared__ alignas(16) __bf16 As[2][GTM * GTK];   // [m][k]
    __shared__ alignas(16) __bf16 Bt[2][GTN * GTK];   // [n][k] transposed tile

    const int tid   = threadIdx.x;
    const int lane  = tid & 31;
    const int wave  = tid >> 5;
    const int waveM = wave >> 2;       // 0..1
    const int waveN = wave & 3;        // 0..3
    const int m0 = blockIdx.y * GTM;
    const int n0 = blockIdx.x * GTN;

    const int kp  = tid >> 4;          // 0..15 : k-row pair for B staging
    const int seg = tid & 15;          // 0..15 : 8-wide n segment

    constexpr int NSTEP = K / GTK;

    v8f zero = {};
    v8f acc[4][2];
#pragma unroll
    for (int i = 0; i < 4; ++i)
#pragma unroll
        for (int j = 0; j < 2; ++j) acc[i][j] = zero;

#ifdef HAVE_TDM
    const unsigned asOff0 = (unsigned)(size_t)(void*)&As[0][0];  // LDS byte offset
#endif

    union BS { uint4 q; unsigned short h[8]; };

    // -------- prologue: stage tile 0 into buffer 0, full wait --------
#ifdef HAVE_TDM
    if (wave == 0)
        tdm_load_2d_to_lds(asOff0, &A[(size_t)m0 * K], GTK * 2 / 8, GTM, K * 2 / 8);
#else
#pragma unroll
    for (int j = 0; j < 2; ++j) {
        int e = tid + j * 256;
        *(uint4*)(&As[0][(e >> 2) * GTK + (e & 3) * 8]) =
            *(const uint4*)(&A[(size_t)(m0 + (e >> 2)) * K + (e & 3) * 8]);
    }
#endif
    {
        BS b0, b1;
        b0.q = *(const uint4*)(&B[(size_t)(2 * kp)     * N + n0 + seg * 8]);
        b1.q = *(const uint4*)(&B[(size_t)(2 * kp + 1) * N + n0 + seg * 8]);
        unsigned* bt32 = (unsigned*)&Bt[0][0];
#pragma unroll
        for (int j = 0; j < 8; ++j)
            bt32[(seg * 8 + j) * (GTK / 2) + kp] =
                (unsigned)b0.h[j] | ((unsigned)b1.h[j] << 16);
    }
#ifdef HAVE_TDM
    if (wave == 0) __builtin_amdgcn_s_wait_tensorcnt(0);
#endif
    __syncthreads();

    // -------- main loop: compute tile i from cur, prefetch tile i+1 --------
    for (int i = 0; i < NSTEP; ++i) {
        const int cur = i & 1;
        const int nxt = cur ^ 1;
        const int k1  = (i + 1) * GTK;
        const bool havePrefetch = (i + 1 < NSTEP);

        BS b0, b1;
#ifndef HAVE_TDM
        uint4 ar[2];
#endif
        if (havePrefetch) {
#ifdef HAVE_TDM
            if (wave == 0)
                tdm_load_2d_to_lds(asOff0 + (unsigned)(nxt * GTM * GTK * 2),
                                   &A[(size_t)m0 * K + k1],
                                   GTK * 2 / 8, GTM, K * 2 / 8);
#else
#pragma unroll
            for (int j = 0; j < 2; ++j) {
                int e = tid + j * 256;
                ar[j] = *(const uint4*)(&A[(size_t)(m0 + (e >> 2)) * K + k1 + (e & 3) * 8]);
            }
#endif
            b0.q = *(const uint4*)(&B[(size_t)(k1 + 2 * kp)     * N + n0 + seg * 8]);
            b1.q = *(const uint4*)(&B[(size_t)(k1 + 2 * kp + 1) * N + n0 + seg * 8]);
        }

        // ---- compute tile i (overlaps TDM + B loads in flight) ----
        v16bf aF[4];
#pragma unroll
        for (int mt = 0; mt < 4; ++mt)
            aF[mt] = load_frag(&As[cur][(waveM * 64 + mt * 16) * GTK], GTK, lane);
#pragma unroll
        for (int nt = 0; nt < 2; ++nt) {
            v16bf bF = load_frag(&Bt[cur][(waveN * 32 + nt * 16) * GTK], GTK, lane);
#pragma unroll
            for (int mt = 0; mt < 4; ++mt)
                acc[mt][nt] = __builtin_amdgcn_wmma_f32_16x16x32_bf16(
                    false, aF[mt], false, bF, (short)0, acc[mt][nt], false, false);
        }

        // ---- drain prefetch into the alternate buffers ----
        if (havePrefetch) {
#ifndef HAVE_TDM
#pragma unroll
            for (int j = 0; j < 2; ++j) {
                int e = tid + j * 256;
                *(uint4*)(&As[nxt][(e >> 2) * GTK + (e & 3) * 8]) = ar[j];
            }
#endif
            unsigned* bt32 = (unsigned*)&Bt[nxt][0];
#pragma unroll
            for (int j = 0; j < 8; ++j)
                bt32[(seg * 8 + j) * (GTK / 2) + kp] =
                    (unsigned)b0.h[j] | ((unsigned)b1.h[j] << 16);
#ifdef HAVE_TDM
            if (wave == 0) __builtin_amdgcn_s_wait_tensorcnt(0);
#endif
        }
        __syncthreads();
    }

    // Epilogue: one base pointer + compile-time offsets (N is constexpr).
    const int lh   = lane >> 4;
    const int ncol = lane & 15;
    float* cb = C + (size_t)(m0 + waveM * 64 + lh * 8) * N + (n0 + waveN * 32 + ncol);
#pragma unroll
    for (int mt = 0; mt < 4; ++mt)
#pragma unroll
        for (int nt = 0; nt < 2; ++nt)
#pragma unroll
            for (int r = 0; r < 8; ++r)
                cb[(size_t)(mt * 16 + r) * N + nt * 16] = acc[mt][nt][r];
}

// ---------------------------------------------------------------------------
// RoPE repack for Q: Q32 [B*T, NH*HD] f32 -> Qb [B, NH, T, HD] bf16 (rotated)
// ---------------------------------------------------------------------------
__global__ void rope_q_kernel(const float* __restrict__ Q32, __bf16* __restrict__ Qb) {
    int idx = blockIdx.x * blockDim.x + threadIdx.x;
    if (idx >= CB * CT * CNH * 64) return;
    int i = idx & 63;
    int h = (idx >> 6) & 15;
    int t = (idx >> 10) & 2047;
    int b = idx >> 21;
    size_t row = (size_t)(b * CT + t);
    float q0 = Q32[row * CHID + h * CHD + i];
    float q1 = Q32[row * CHID + h * CHD + i + 64];
    float ang = (float)t * __expf(-(float)i * 0.14391156f);  // ln(1e4)/64
    float c = __cosf(ang), s = __sinf(ang);
    size_t ob = ((size_t)(b * CNH + h) * CT + t) * CHD;
    Qb[ob + i]      = (__bf16)(q0 * c - q1 * s);
    Qb[ob + i + 64] = (__bf16)(q1 * c + q0 * s);
}

// ---------------------------------------------------------------------------
// RoPE K + transposed V repack:
//   K32 -> Kb [B, NKV, T, HD] bf16 (rotated);  V32 -> Vtb [B, NKV, HD, T] bf16
// ---------------------------------------------------------------------------
__global__ void rope_kv_kernel(const float* __restrict__ K32,
                               const float* __restrict__ V32,
                               __bf16* __restrict__ Kb, __bf16* __restrict__ Vtb) {
    int idx = blockIdx.x * blockDim.x + threadIdx.x;
    if (idx >= CB * CT * CNKV * 64) return;
    int i  = idx & 63;
    int kh = (idx >> 6) & 3;
    int t  = (idx >> 8) & 2047;
    int b  = idx >> 19;
    size_t row = (size_t)(b * CT + t);
    float k0 = K32[row * CKVD + kh * CHD + i];
    float k1 = K32[row * CKVD + kh * CHD + i + 64];
    float ang = (float)t * __expf(-(float)i * 0.14391156f);
    float c = __cosf(ang), s = __sinf(ang);
    size_t kb = ((size_t)(b * CNKV + kh) * CT + t) * CHD;
    Kb[kb + i]      = (__bf16)(k0 * c - k1 * s);
    Kb[kb + i + 64] = (__bf16)(k1 * c + k0 * s);

    float v0 = V32[row * CKVD + kh * CHD + i];
    float v1 = V32[row * CKVD + kh * CHD + i + 64];
    size_t vb = (size_t)(b * CNKV + kh) * CHD;
    Vtb[(vb + i)      * CT + t] = (__bf16)v0;
    Vtb[(vb + i + 64) * CT + t] = (__bf16)v1;
}

// ---------------------------------------------------------------------------
// Flash attention (causal, GQA).  128 threads = 4 waves; wave owns 16 q-rows.
// 32-wide KV chunks: 8 WMMA scores, online softmax (16-lane shfl reductions),
// P relayout through per-wave LDS, 8 WMMA for P@V against transposed V.
// ---------------------------------------------------------------------------
__global__ __launch_bounds__(128)
void flash_attn_kernel(const __bf16* __restrict__ Qb, const __bf16* __restrict__ Kb,
                       const __bf16* __restrict__ Vtb, __bf16* __restrict__ Ob) {
    __shared__ alignas(16) __bf16 Pbuf[4][16 * 32];

    const int lane = threadIdx.x & 31;
    const int wave = threadIdx.x >> 5;
    const int bh   = blockIdx.x;
    const int b    = bh >> 4;
    const int h    = bh & 15;
    const int kvh  = h >> 2;
    const int q0   = (blockIdx.y * 4 + wave) * 16;

    const __bf16* Qbase = Qb  + ((size_t)(b * CNH  + h)   * CT + q0) * CHD;
    const __bf16* Kbase = Kb  + ((size_t)(b * CNKV + kvh) * CT) * CHD;
    const __bf16* Vbase = Vtb + ((size_t)(b * CNKV + kvh) * CHD) * CT;

    v16bf qF[4];
#pragma unroll
    for (int kd = 0; kd < 4; ++kd)
        qF[kd] = load_frag(Qbase + kd * 32, CHD, lane);

    v8f zero = {};
    v8f o[8];
#pragma unroll
    for (int i = 0; i < 8; ++i) o[i] = zero;
    float mrow[8], lrow[8];
#pragma unroll
    for (int r = 0; r < 8; ++r) { mrow[r] = -3.0e38f; lrow[r] = 0.0f; }

    const int lh   = lane >> 4;
    const int ncol = lane & 15;
    const float scale = 0.08838834764831845f;   // 1/sqrt(128)

    const int nChunks = (q0 >> 5) + 1;
    for (int c = 0; c < nChunks; ++c) {
        const int kpos0 = c * 32;

        v8f s0 = zero, s1 = zero;
#pragma unroll
        for (int kd = 0; kd < 4; ++kd) {
            v16bf kf = load_frag(Kbase + (size_t)kpos0 * CHD + kd * 32, CHD, lane);
            s0 = __builtin_amdgcn_wmma_f32_16x16x32_bf16(
                false, qF[kd], false, kf, (short)0, s0, false, false);
        }
#pragma unroll
        for (int kd = 0; kd < 4; ++kd) {
            v16bf kf = load_frag(Kbase + (size_t)(kpos0 + 16) * CHD + kd * 32, CHD, lane);
            s1 = __builtin_amdgcn_wmma_f32_16x16x32_bf16(
                false, qF[kd], false, kf, (short)0, s1, false, false);
        }

        float alpha[8];
#pragma unroll
        for (int r = 0; r < 8; ++r) {
            int qg = q0 + r + 8 * lh;
            float v0 = s0[r] * scale;
            float v1 = s1[r] * scale;
            if (kpos0 + ncol      > qg) v0 = -1.0e30f;
            if (kpos0 + 16 + ncol > qg) v1 = -1.0e30f;
            float rm = fmaxf(v0, v1);
#pragma unroll
            for (int mk = 1; mk < 16; mk <<= 1)
                rm = fmaxf(rm, __shfl_xor(rm, mk, 32));
            float mnew = fmaxf(mrow[r], rm);
            float a  = __expf(mrow[r] - mnew);
            float p0 = __expf(v0 - mnew);
            float p1 = __expf(v1 - mnew);
            float rs = p0 + p1;
#pragma unroll
            for (int mk = 1; mk < 16; mk <<= 1)
                rs += __shfl_xor(rs, mk, 32);
            lrow[r] = lrow[r] * a + rs;
            mrow[r] = mnew;
            alpha[r] = a;
            Pbuf[wave][(r + 8 * lh) * 32 + ncol]      = (__bf16)p0;
            Pbuf[wave][(r + 8 * lh) * 32 + 16 + ncol] = (__bf16)p1;
        }

#pragma unroll
        for (int dg = 0; dg < 8; ++dg)
#pragma unroll
            for (int r = 0; r < 8; ++r)
                o[dg][r] *= alpha[r];

        asm volatile("s_wait_dscnt 0" ::: "memory");
        v16bf pF = load_frag(&Pbuf[wave][0], 32, lane);
        asm volatile("s_wait_dscnt 0" ::: "memory");

#pragma unroll
        for (int dg = 0; dg < 8; ++dg) {
            v16bf vf = load_frag(Vbase + (size_t)(dg * 16) * CT + kpos0, CT, lane);
            o[dg] = __builtin_amdgcn_wmma_f32_16x16x32_bf16(
                false, pF, false, vf, (short)0, o[dg], false, false);
        }
    }

    __bf16* obase = Ob + ((size_t)(b * CT + q0)) * (CNH * CHD) + h * CHD;
#pragma unroll
    for (int r = 0; r < 8; ++r) {
        float inv = 1.0f / lrow[r];
        int m = r + 8 * lh;
#pragma unroll
        for (int dg = 0; dg < 8; ++dg)
            obase[(size_t)m * (CNH * CHD) + dg * 16 + ncol] = (__bf16)(o[dg][r] * inv);
    }
}

// ---------------------------------------------------------------------------
// Host-side launcher
// ---------------------------------------------------------------------------
extern "C" void kernel_launch(void* const* d_in, const int* in_sizes, int n_in,
                              void* d_out, int out_size, void* d_ws, size_t ws_size,
                              hipStream_t stream) {
    const float* x  = (const float*)d_in[0];
    const float* Wq = (const float*)d_in[1];
    const float* Wk = (const float*)d_in[2];
    const float* Wv = (const float*)d_in[3];
    const float* Wo = (const float*)d_in[4];
    float* out = (float*)d_out;

    char* ws = (char*)d_ws;
    size_t off = 0;
    auto alloc = [&](size_t bytes) -> char* {
        char* p = ws + off;
        off = (off + bytes + 255) & ~(size_t)255;
        return p;
    };

    __bf16* xb  = (__bf16*)alloc((size_t)CM * CHID * 2);
    __bf16* Wqb = (__bf16*)alloc((size_t)CHID * CHID * 2);
    __bf16* Wkb = (__bf16*)alloc((size_t)CHID * CKVD * 2);
    __bf16* Wvb = (__bf16*)alloc((size_t)CHID * CKVD * 2);
    __bf16* Wob = (__bf16*)alloc((size_t)CHID * CHID * 2);
    float*  Q32 = (float*)alloc((size_t)CM * CHID * 4);
    float*  K32 = (float*)alloc((size_t)CM * CKVD * 4);
    float*  V32 = (float*)alloc((size_t)CM * CKVD * 4);
    __bf16* Qbv = (__bf16*)alloc((size_t)CM * CHID * 2);
    __bf16* Kbv = (__bf16*)alloc((size_t)CM * CKVD * 2);
    __bf16* Vtb = (__bf16*)alloc((size_t)CM * CKVD * 2);
    __bf16* Ob  = (__bf16*)alloc((size_t)CM * CHID * 2);

    const int THR = 256;
    auto cvt = [&](const float* src, __bf16* dst, int n) {
        f32_to_bf16_kernel<<<(n + THR - 1) / THR, THR, 0, stream>>>(src, dst, n);
    };
    cvt(x,  xb,  CM * CHID);
    cvt(Wq, Wqb, CHID * CHID);
    cvt(Wk, Wkb, CHID * CKVD);
    cvt(Wv, Wvb, CHID * CKVD);
    cvt(Wo, Wob, CHID * CHID);

    // QKV projections
    gemm_bf16_kernel<CHID, CHID><<<dim3(CHID / GTN, CM / GTM), 256, 0, stream>>>(
        xb, Wqb, Q32);
    gemm_bf16_kernel<CKVD, CHID><<<dim3(CKVD / GTN, CM / GTM), 256, 0, stream>>>(
        xb, Wkb, K32);
    gemm_bf16_kernel<CKVD, CHID><<<dim3(CKVD / GTN, CM / GTM), 256, 0, stream>>>(
        xb, Wvb, V32);

    // RoPE + repack
    rope_q_kernel<<<(CB * CT * CNH * 64) / THR, THR, 0, stream>>>(Q32, Qbv);
    rope_kv_kernel<<<(CB * CT * CNKV * 64) / THR, THR, 0, stream>>>(K32, V32, Kbv, Vtb);

    // Flash attention: grid (B*NH, T / (16 rows * 4 waves))
    flash_attn_kernel<<<dim3(CB * CNH, CT / 64), 128, 0, stream>>>(Qbv, Kbv, Vtb, Ob);

    // Output projection -> f32 out
    gemm_bf16_kernel<CHID, CHID><<<dim3(CHID / GTN, CM / GTM), 256, 0, stream>>>(
        Ob, Wob, out);
}
